// Attention_6871947674089
// MI455X (gfx1250) — compile-verified, matrix operand order checked by reading, not run on hardware
//
#include <hip/hip_runtime.h>
#include <hip/hip_bf16.h>
#include <math.h>
#include <stdint.h>

typedef __attribute__((ext_vector_type(16))) _Float16 v16h;
typedef __attribute__((ext_vector_type(8)))  _Float16 v8h;
typedef __attribute__((ext_vector_type(8)))  float    v8f;
typedef __attribute__((ext_vector_type(4)))  unsigned int v4u;
typedef __attribute__((ext_vector_type(8)))  int      v8i;
typedef __attribute__((ext_vector_type(4)))  int      v4i;

#define WMMA_F16(a,b,c) __builtin_amdgcn_wmma_f32_16x16x32_f16( \
    false,(a),false,(b),(short)0,(c),false,false)

static __device__ inline v16h pack16(v8h lo, v8h hi) {
    v16h r;
#pragma unroll
    for (int i = 0; i < 8; ++i) { r[i] = lo[i]; r[i + 8] = hi[i]; }
    return r;
}

// One lane's 32-element K-run in WMMA A/B register layout (16-bit data):
//   e=0..7  -> rowptr[khalf*8 + e]
//   e=8..15 -> rowptr[16 + khalf*8 + (e-8)]
static __device__ inline v16h load_run_f16(const _Float16* rowptr, int khalf) {
    v8h lo = *(const v8h*)(rowptr + khalf * 8);
    v8h hi = *(const v8h*)(rowptr + 16 + khalf * 8);
    return pack16(lo, hi);
}

static __device__ inline v16h load_run_f32(const float* rowptr, int khalf) {
    v8h lo, hi;
#pragma unroll
    for (int i = 0; i < 8; ++i) {
        lo[i] = (_Float16)rowptr[khalf * 8 + i];
        hi[i] = (_Float16)rowptr[16 + khalf * 8 + i];
    }
    return pack16(lo, hi);
}

// ---- Tensor Data Mover: 2D f16 tile (rows x cols, row stride in elems) -> LDS ----
// D# per CDNA5 ISA 8.3/8.4. tensor dims == tile dims (no OOB clipping needed).
static __device__ inline void tdm_load_2d_f16(unsigned lds_off, const _Float16* gptr,
                                              unsigned rows, unsigned cols,
                                              unsigned stride_elems) {
    unsigned long long ga = (unsigned long long)(uintptr_t)gptr;
    v4u g0 = { 1u,                                   // count=1, user mode
               lds_off,                              // lds_addr (bytes)
               (unsigned)ga,                         // global_addr[31:0]
               (unsigned)(ga >> 32) | (2u << 30) };  // global_addr[56:32] | type=2
    v8i g1 = { (int)(1u << 16),                      // data_size=1 (2 bytes)
               (int)(cols << 16),                    // tensor_dim0[15:0] in [31:16]
               (int)(rows << 16),                    // tensor_dim1[15:0] in [31:16]
               (int)(cols << 16),                    // tile_dim0 in [31:16]
               (int)rows,                            // tile_dim1 ([15:0]); tile_dim2=0
               (int)stride_elems,                    // tensor_dim0_stride[31:0]
               0, 0 };                               // dim1_stride unused (2D)
    v4i z4 = {0, 0, 0, 0};
#if defined(__clang_major__) && (__clang_major__ >= 23)
    v8i z8 = {0, 0, 0, 0, 0, 0, 0, 0};
    __builtin_amdgcn_tensor_load_to_lds(g0, g1, z4, z4, z8, 0);
#else
    __builtin_amdgcn_tensor_load_to_lds(g0, g1, z4, z4, 0);
#endif
}

// -------- Kernel 1: X(f32)[M x Kd] @ W(f32)[Kd x N] -> Y(f16)[M x N] * scale --------
__global__ __launch_bounds__(256) void proj_f32_to_f16(
    const float* __restrict__ X, const float* __restrict__ W,
    _Float16* __restrict__ Y, int Kd, int N, float scale)
{
    const int lane  = threadIdx.x & 31;
    const int khalf = lane >> 4;
    const int gw    = (blockIdx.x * blockDim.x + threadIdx.x) >> 5;
    const int ntil  = N >> 4;
    const int tm = gw / ntil, tn = gw % ntil;
    const int arow = tm * 16 + (lane & 15);
    const int bcol = tn * 16 + (lane & 15);
    v8f c = {};
    for (int k0 = 0; k0 < Kd; k0 += 32) {
        v16h a = load_run_f32(X + (size_t)arow * Kd + k0, khalf);
        v16h b;
#pragma unroll
        for (int e = 0; e < 16; ++e) {
            int kk = k0 + ((e >> 3) << 4) + khalf * 8 + (e & 7);
            b[e] = (_Float16)W[(size_t)kk * N + bcol];
        }
        c = WMMA_F16(a, b, c);
    }
#pragma unroll
    for (int r = 0; r < 8; ++r) {
        int orow = tm * 16 + khalf * 8 + r;
        Y[(size_t)orow * N + bcol] = (_Float16)(c[r] * scale);
    }
}

// -------- Kernel 3: X(f16)[M x Kd] @ W(f32)[Kd x N] + bias -> Y(f32)[M x N] --------
__global__ __launch_bounds__(256) void proj_f16_to_f32_bias(
    const _Float16* __restrict__ X, const float* __restrict__ W,
    const float* __restrict__ bias, float* __restrict__ Y, int Kd, int N)
{
    const int lane  = threadIdx.x & 31;
    const int khalf = lane >> 4;
    const int gw    = (blockIdx.x * blockDim.x + threadIdx.x) >> 5;
    const int ntil  = N >> 4;
    const int tm = gw / ntil, tn = gw % ntil;
    const int arow = tm * 16 + (lane & 15);
    const int bcol = tn * 16 + (lane & 15);
    v8f c = {};
    for (int k0 = 0; k0 < Kd; k0 += 32) {
        v16h a = load_run_f16(X + (size_t)arow * Kd + k0, khalf);
        v16h b;
#pragma unroll
        for (int e = 0; e < 16; ++e) {
            int kk = k0 + ((e >> 3) << 4) + khalf * 8 + (e & 7);
            b[e] = (_Float16)W[(size_t)kk * N + bcol];
        }
        c = WMMA_F16(a, b, c);
    }
    const float bb = bias[bcol];
#pragma unroll
    for (int r = 0; r < 8; ++r) {
        int orow = tm * 16 + khalf * 8 + r;
        Y[(size_t)orow * N + bcol] = c[r] + bb;
    }
}

// -------- Kernel 2: attention. Block = 8 waves sharing one (b,h); wave = 16 q rows.
// K/V tiles staged to LDS once per block via the Tensor Data Mover (TENSORcnt).
__global__ __launch_bounds__(256) void attn_kernel(
    const _Float16* __restrict__ qws, const _Float16* __restrict__ kws,
    const _Float16* __restrict__ vws, _Float16* __restrict__ ows,
    float* __restrict__ wgt)
{
    __shared__ __align__(16) _Float16 kbuf[32 * 32];   // up to 32 keys x 32 d
    __shared__ __align__(16) _Float16 vbuf[32 * 32];   // 32 keys x 32 d
    __shared__ __align__(16) _Float16 pbuf[8][16 * 32];

    const int lane  = threadIdx.x & 31;
    const int khalf = lane >> 4;
    const int wv    = threadIdx.x >> 5;

    const int b  = blockIdx.x >> 8;          // 256 blocks per batch
    const int h  = (blockIdx.x >> 5) & 7;    // 32 blocks per head
    const int qg = blockIdx.x & 31;
    const int qt = qg * 8 + wv;              // this wave's 16-row q tile

    const size_t seq = 4096, dm = 256;
    const _Float16* qh = qws + ((size_t)b * seq + (size_t)qt * 16) * dm + h * 32;
    const _Float16* kh = kws + (size_t)b * seq * dm + h * 32;
    const _Float16* vh = vws + (size_t)b * seq * dm + h * 32;

    const unsigned koff = (unsigned)(uintptr_t)(void*)kbuf;
    const unsigned voff = (unsigned)(uintptr_t)(void*)vbuf;

    // A operand: 16x32 q tile (1/sqrt(d) folded in at projection), kept for the kernel
    const v16h aq = load_run_f16(qh + (size_t)(lane & 15) * dm, khalf);

    float m[8], l[8];
#pragma unroll
    for (int r = 0; r < 8; ++r) { m[r] = -3.0e38f; l[r] = 0.f; }

    // ---- Pass 1: online row-max / row-sum over all 4096 keys ----
    for (int kt = 0; kt < 256; ++kt) {
        if (wv == 0) {
            tdm_load_2d_f16(koff, kh + (size_t)kt * 16 * dm, 16, 32, (unsigned)dm);
            __builtin_amdgcn_s_wait_tensorcnt(0);
        }
        __syncthreads();
        v16h bk = load_run_f16(kbuf + (size_t)(lane & 15) * 32, khalf);
        v8f s = {};
        s = WMMA_F16(aq, bk, s);
#pragma unroll
        for (int r = 0; r < 8; ++r) {
            float t = s[r];
            t = fmaxf(t, __shfl_xor(t, 1, 32));
            t = fmaxf(t, __shfl_xor(t, 2, 32));
            t = fmaxf(t, __shfl_xor(t, 4, 32));
            t = fmaxf(t, __shfl_xor(t, 8, 32));
            float mn = fmaxf(m[r], t);
            float e = __expf(s[r] - mn);
            e += __shfl_xor(e, 1, 32);
            e += __shfl_xor(e, 2, 32);
            e += __shfl_xor(e, 4, 32);
            e += __shfl_xor(e, 8, 32);
            l[r] = l[r] * __expf(m[r] - mn) + e;
            m[r] = mn;
        }
        __syncthreads();   // protect kbuf before next TDM overwrite
    }
    float il[8];
#pragma unroll
    for (int r = 0; r < 8; ++r) il[r] = 1.f / l[r];

    // ---- Pass 2: recompute scores, write normalized weights once, accumulate O ----
    v8f o0 = {}, o1 = {};
    float* wrow = wgt + (((size_t)(b * 8 + h)) * seq + (size_t)qt * 16) * seq;
    for (int kp = 0; kp < 128; ++kp) {
        if (wv == 0) {
            tdm_load_2d_f16(koff, kh + (size_t)kp * 32 * dm, 32, 32, (unsigned)dm);
            tdm_load_2d_f16(voff, vh + (size_t)kp * 32 * dm, 32, 32, (unsigned)dm);
            __builtin_amdgcn_s_wait_tensorcnt(0);
        }
        __syncthreads();
#pragma unroll
        for (int t = 0; t < 2; ++t) {
            const int kt = kp * 2 + t;
            v16h bk = load_run_f16(kbuf + (size_t)(t * 16 + (lane & 15)) * 32, khalf);
            v8f s = {};
            s = WMMA_F16(aq, bk, s);
#pragma unroll
            for (int r = 0; r < 8; ++r) {
                const int lr = khalf * 8 + r;                 // local row 0..15
                float p = __expf(s[r] - m[r]) * il[r];
                wrow[(size_t)lr * seq + kt * 16 + (lane & 15)] = p;
                pbuf[wv][lr * 32 + t * 16 + (lane & 15)] = (_Float16)p;
            }
        }
        // reload P (16 rows x 32 keys) from LDS in WMMA-A layout
        v16h ap = load_run_f16(&pbuf[wv][(lane & 15) * 32], khalf);
#pragma unroll
        for (int n = 0; n < 2; ++n) {
            v16h bv;
#pragma unroll
            for (int e = 0; e < 16; ++e) {
                int key = ((e >> 3) << 4) + khalf * 8 + (e & 7);
                bv[e] = vbuf[key * 32 + n * 16 + (lane & 15)];
            }
            if (n == 0) o0 = WMMA_F16(ap, bv, o0);
            else        o1 = WMMA_F16(ap, bv, o1);
        }
        __syncthreads();   // protect kbuf/vbuf before next TDM overwrite
    }

    // store O tile (f16) into inner workspace [B*Q x 256]
#pragma unroll
    for (int r = 0; r < 8; ++r) {
        size_t orow = (size_t)b * seq + (size_t)qt * 16 + khalf * 8 + r;
        ows[orow * dm + h * 32 + (lane & 15)]      = (_Float16)o0[r];
        ows[orow * dm + h * 32 + 16 + (lane & 15)] = (_Float16)o1[r];
    }
}

extern "C" void kernel_launch(void* const* d_in, const int* in_sizes, int n_in,
                              void* d_out, int out_size, void* d_ws, size_t ws_size,
                              hipStream_t stream)
{
    const float* query = (const float*)d_in[0];
    const float* key_  = (const float*)d_in[1];
    const float* value = (const float*)d_in[2];
    const float* Wq    = (const float*)d_in[3];
    const float* Wk    = (const float*)d_in[4];
    const float* Wv    = (const float*)d_in[5];
    const float* Wo    = (const float*)d_in[6];
    const float* bo    = (const float*)d_in[7];

    const int B = 2, Q = 4096, Dm = 512, Inner = 256;
    const int M = B * Q;  // 8192

    char* ws = (char*)d_ws;
    const size_t seg = (size_t)M * Inner * sizeof(_Float16); // 4 MB each
    _Float16* qws = (_Float16*)(ws + 0 * seg);
    _Float16* kws = (_Float16*)(ws + 1 * seg);
    _Float16* vws = (_Float16*)(ws + 2 * seg);
    _Float16* ows = (_Float16*)(ws + 3 * seg);

    float* attn_out = (float*)d_out;                    // [B,Q,512]
    float* attn_wgt = attn_out + (size_t)M * Dm;        // [B,H,Q,K]

    const float scale = 0.17677669529663687f;           // 1/sqrt(32)

    // Projections: (M/16)*(Inner/16) = 8192 tiles, 8 waves/block -> 1024 blocks
    {
        dim3 blk(256), grd((M / 16) * (Inner / 16) / 8);
        proj_f32_to_f16<<<grd, blk, 0, stream>>>(query, Wq, qws, Dm, Inner, scale);
        proj_f32_to_f16<<<grd, blk, 0, stream>>>(key_,  Wk, kws, Dm, Inner, 1.0f);
        proj_f32_to_f16<<<grd, blk, 0, stream>>>(value, Wv, vws, Dm, Inner, 1.0f);
    }
    // Attention: 512 blocks of 8 waves; each block owns one (b,h) slice of 128 q rows
    {
        dim3 blk(256), grd(512);
        attn_kernel<<<grd, blk, 0, stream>>>(qws, kws, vws, ows, attn_wgt);
    }
    // Output projection: (M/16)*(Dm/16) = 16384 tiles -> 2048 blocks
    {
        dim3 blk(256), grd((M / 16) * (Dm / 16) / 8);
        proj_f16_to_f32_bias<<<grd, blk, 0, stream>>>(ows, Wo, bo, attn_out, Inner, Dm);
    }
}
